// CausalSelfAttentionWithROPE_78726750535686
// MI455X (gfx1250) — matrix operand order repaired and checked by reading, not compiled, not run on hardware
//
#include <hip/hip_runtime.h>
#include <hip/hip_bf16.h>

// ---------------------------------------------------------------------------
// CDNA5 (gfx1250) causal self-attention with RoPE.
//   B=2, T=2048, C=1024, H=16, Dh=64
// Pipeline:
//   0) cvt / transpose prep: x -> bf16 [M,K]; Wqkv,Wproj -> bf16 [N,K]
//   1) gemm_bias_bf16 : qkv = x @ Wqkv + bqkv  (bf16 WMMA, fp32 acc,
//                       double-buffered LDS, async global->LDS staging)
//   2) rope_pack      : fp32 qkv -> bf16 q,k (RoPE'd) [B,H,T,D], vT [B,H,D,T]
//   3) flash_attn     : online-softmax attention, WMMA for QK^T and P@V
//   4) gemm_bias_bf16 : out = y @ Wproj + bproj
// ---------------------------------------------------------------------------

typedef __attribute__((ext_vector_type(8)))  float  v8f;
typedef __attribute__((ext_vector_type(16))) __bf16 v16bf;
typedef __attribute__((ext_vector_type(8)))  __bf16 v8bf;
typedef int b128i __attribute__((vector_size(16)));

#define AS1 __attribute__((address_space(1)))
#define AS3 __attribute__((address_space(3)))

#define BATCH 2
#define SEQ   2048
#define DMODEL 1024
#define NHEAD 16
#define HDIM  64

// ---- CDNA5 async global->LDS path (guarded; falls back to sync copies) ----
#if defined(__gfx1250__) && defined(__has_builtin)
#if __has_builtin(__builtin_amdgcn_global_load_async_to_lds_b128) && \
    __has_builtin(__builtin_amdgcn_s_wait_asynccnt)
#define USE_ASYNC_LDS 1
#endif
#endif
#ifndef USE_ASYNC_LDS
#define USE_ASYNC_LDS 0
#endif

static __device__ __forceinline__ void cp16(__bf16* lds, const __bf16* g) {
#if USE_ASYNC_LDS
  __builtin_amdgcn_global_load_async_to_lds_b128(
      (AS1 b128i*)(AS1 void*)(void*)g,
      (AS3 b128i*)(AS3 void*)lds, 0, 0);
#else
  *(v8bf*)lds = *(const v8bf*)g;
#endif
}

template <int N>
static __device__ __forceinline__ void async_wait_tail() {
#if USE_ASYNC_LDS
  __builtin_amdgcn_s_wait_asynccnt(N);
#endif
}

static __device__ __forceinline__ v16bf bf_cat(v8bf lo, v8bf hi) {
  return __builtin_shufflevector(lo, hi, 0,1,2,3,4,5,6,7,8,9,10,11,12,13,14,15);
}

static __device__ __forceinline__ v8f wmma_bf16(v16bf a, v16bf b, v8f c) {
  return __builtin_amdgcn_wmma_f32_16x16x32_bf16(false, a, false, b, (short)0, c,
                                                 false, false);
}

// ---------------------------------------------------------------------------
// Prep: fp32 -> bf16 copy
// ---------------------------------------------------------------------------
__global__ __launch_bounds__(256)
void cvt_bf16(const float* __restrict__ X, __bf16* __restrict__ Xb, int n) {
  const int i = blockIdx.x * 256 + threadIdx.x;
  if (i < n) Xb[i] = (__bf16)X[i];
}

// Prep: W [K,N] fp32 -> WT [N,K] bf16 (k fastest in output => coalesced writes)
__global__ __launch_bounds__(256)
void transpose_cvt(const float* __restrict__ W, __bf16* __restrict__ WT,
                   int K, int N) {
  const size_t idx = (size_t)blockIdx.x * 256 + threadIdx.x;
  if (idx >= (size_t)K * N) return;
  const int k = (int)(idx % K);
  const int n = (int)(idx / K);
  WT[idx] = (__bf16)W[(size_t)k * N + n];
}

// ---------------------------------------------------------------------------
// GEMM: C[M,N] = A[M,K](bf16) * BT[N,K](bf16, pre-transposed) + bias[N]
// Block 256 threads = 8 waves (4 M-waves x 2 N-waves).
// Block tile 128x128, K step 32, double-buffered LDS, async staging.
// Wave tile 32x64 => 8 WMMA per K step.
// ---------------------------------------------------------------------------
__global__ __launch_bounds__(256)
void gemm_bias_bf16(const __bf16* __restrict__ A, const __bf16* __restrict__ BT,
                    const float* __restrict__ bias, float* __restrict__ Cout,
                    int M, int N, int K) {
  __shared__ __align__(32) __bf16 sA[2][128][32];
  __shared__ __align__(32) __bf16 sB[2][128][32];

  const int tid  = threadIdx.x;
  const int lane = tid & 31;
  const int wave = tid >> 5;
  const int r    = lane & 15;
  const int half = lane >> 4;
  const int wm   = (wave >> 1) * 32;   // wave M offset in block tile
  const int wn   = (wave & 1) * 64;    // wave N offset in block tile

  const int bm = blockIdx.y;
  const int bn = blockIdx.x;

  v8f acc[2][4];
#pragma unroll
  for (int i = 0; i < 2; ++i)
#pragma unroll
    for (int j = 0; j < 4; ++j) acc[i][j] = (v8f)0.0f;

  const __bf16* Ab = A  + (size_t)(bm * 128) * K;
  const __bf16* Bb = BT + (size_t)(bn * 128) * K;

  const int srow = tid >> 2;        // 0..63
  const int scol = (tid & 3) * 8;   // 0,8,16,24

  // stage one 128x32 A tile + 128x32 B tile into buffer `buf`
  auto stage = [&](int buf, int kb) {
    cp16(&sA[buf][srow][scol],      Ab + (size_t)srow * K + kb + scol);
    cp16(&sA[buf][srow + 64][scol], Ab + (size_t)(srow + 64) * K + kb + scol);
    cp16(&sB[buf][srow][scol],      Bb + (size_t)srow * K + kb + scol);
    cp16(&sB[buf][srow + 64][scol], Bb + (size_t)(srow + 64) * K + kb + scol);
  };

  const int nt = K / 32;
  stage(0, 0);

  for (int kt = 0; kt < nt; ++kt) {
    const int buf = kt & 1;
    if (kt + 1 < nt) {
      stage(buf ^ 1, (kt + 1) * 32);
      async_wait_tail<4>();   // 4 in-flight copies belong to the next tile
    } else {
      async_wait_tail<0>();
    }
    __syncthreads();

    v16bf af[2];
#pragma unroll
    for (int mi = 0; mi < 2; ++mi) {
      const __bf16* p = &sA[buf][wm + mi * 16 + r][half * 16];
      af[mi] = bf_cat(*(const v8bf*)p, *(const v8bf*)(p + 8));
    }
    v16bf bf[4];
#pragma unroll
    for (int ni = 0; ni < 4; ++ni)
      bf[ni] = *(const v16bf*)&sB[buf][wn + ni * 16 + r][half * 16];

#pragma unroll
    for (int mi = 0; mi < 2; ++mi)
#pragma unroll
      for (int ni = 0; ni < 4; ++ni)
        acc[mi][ni] = wmma_bf16(af[mi], bf[ni], acc[mi][ni]);

    __syncthreads();
  }

  // epilogue: bias + fp32 store
#pragma unroll
  for (int ni = 0; ni < 4; ++ni) {
    const int gn = bn * 128 + wn + ni * 16 + r;
    const float bv = bias[gn];
#pragma unroll
    for (int mi = 0; mi < 2; ++mi) {
#pragma unroll
      for (int j = 0; j < 8; ++j) {
        const int gm = bm * 128 + wm + mi * 16 + half * 8 + j;
        Cout[(size_t)gm * N + gn] = acc[mi][ni][j] + bv;
      }
    }
  }
}

// ---------------------------------------------------------------------------
// RoPE + pack: qkv fp32 [B,T,3C] -> q,k bf16 [B,H,T,D] (RoPE applied),
//              v bf16 transposed [B,H,D,T].
// ---------------------------------------------------------------------------
__global__ __launch_bounds__(256)
void rope_pack(const float* __restrict__ qkv, __bf16* __restrict__ q,
               __bf16* __restrict__ k, __bf16* __restrict__ vT) {
  const int idx = blockIdx.x * blockDim.x + threadIdx.x;
  if (idx >= BATCH * SEQ * NHEAD * HDIM) return;
  const int d = idx & (HDIM - 1);
  const int h = (idx >> 6) & (NHEAD - 1);
  const int t = (idx >> 10) & (SEQ - 1);
  const int b = idx >> 21;

  const size_t row = (size_t)(b * SEQ + t) * (3 * DMODEL);
  const float vv = qkv[row + 2 * DMODEL + h * HDIM + d];
  vT[((size_t)(b * NHEAD + h) * HDIM + d) * SEQ + t] = (__bf16)vv;

  if (d < HDIM / 2) {
    const int i = d;
    const float theta = __powf(10000.0f, -(2.0f * (float)i) / (float)HDIM);
    const float fr = (float)t * theta;
    float sn, cs;
    __sincosf(fr, &sn, &cs);
    const float q0 = qkv[row + h * HDIM + 2 * i];
    const float q1 = qkv[row + h * HDIM + 2 * i + 1];
    const float k0 = qkv[row + DMODEL + h * HDIM + 2 * i];
    const float k1 = qkv[row + DMODEL + h * HDIM + 2 * i + 1];
    const size_t ob = ((size_t)(b * NHEAD + h) * SEQ + t) * HDIM + 2 * i;
    q[ob]     = (__bf16)(q0 * cs - q1 * sn);
    q[ob + 1] = (__bf16)(q0 * sn + q1 * cs);
    k[ob]     = (__bf16)(k0 * cs - k1 * sn);
    k[ob + 1] = (__bf16)(k0 * sn + k1 * cs);
  }
}

// ---------------------------------------------------------------------------
// Flash attention, causal. Grid: (B*H) * (T/64) blocks, 128 threads = 4 waves.
// Each wave owns 16 query rows; streams 64-key tiles with online softmax.
// Writes y as bf16 [B,T,C] so the projection GEMM consumes it directly.
// ---------------------------------------------------------------------------
__global__ __launch_bounds__(128)
void flash_attn(const __bf16* __restrict__ qr, const __bf16* __restrict__ kr,
                const __bf16* __restrict__ vT, __bf16* __restrict__ y) {
  const int qtiles = SEQ / 64;
  const int bh = blockIdx.x / qtiles;
  const int qt = blockIdx.x % qtiles;
  const int b  = bh / NHEAD;
  const int h  = bh % NHEAD;

  const int lane = threadIdx.x & 31;
  const int wave = threadIdx.x >> 5;
  const int r    = lane & 15;
  const int half = lane >> 4;
  const int q0   = qt * 64 + wave * 16;

  __shared__ __align__(32) __bf16 ldsP[4][16][64];

  const __bf16* qb = qr + ((size_t)bh * SEQ + q0) * HDIM;
  const __bf16* kb = kr + (size_t)bh * SEQ * HDIM;
  const __bf16* vb = vT + (size_t)bh * HDIM * SEQ;

  v16bf qa[2];
#pragma unroll
  for (int ds = 0; ds < 2; ++ds) {
    const __bf16* p = qb + (size_t)r * HDIM + ds * 32 + half * 16;
    qa[ds] = bf_cat(*(const v8bf*)p, *(const v8bf*)(p + 8));
  }

  v8f o[4];
#pragma unroll
  for (int i = 0; i < 4; ++i) o[i] = (v8f)0.0f;
  float mrow[8], lrow[8];
#pragma unroll
  for (int j = 0; j < 8; ++j) { mrow[j] = -1e30f; lrow[j] = 0.0f; }

  const float scale = 0.125f;  // 1/sqrt(64)

  for (int kt = 0; kt <= q0 + 15; kt += 64) {
    v8f s[4];
#pragma unroll
    for (int i = 0; i < 4; ++i) s[i] = (v8f)0.0f;
#pragma unroll
    for (int ds = 0; ds < 2; ++ds) {
#pragma unroll
      for (int nf = 0; nf < 4; ++nf) {
        const int key = kt + nf * 16 + r;
        v16bf kf = *(const v16bf*)(kb + (size_t)key * HDIM + ds * 32 + half * 16);
        s[nf] = wmma_bf16(qa[ds], kf, s[nf]);
      }
    }

    float mloc[8];
#pragma unroll
    for (int j = 0; j < 8; ++j) mloc[j] = -1e30f;
#pragma unroll
    for (int nf = 0; nf < 4; ++nf) {
      const int key = kt + nf * 16 + r;
#pragma unroll
      for (int j = 0; j < 8; ++j) {
        const int qi = q0 + half * 8 + j;
        float sv = s[nf][j] * scale;
        sv = (key <= qi) ? sv : -1e30f;
        s[nf][j] = sv;
        mloc[j] = fmaxf(mloc[j], sv);
      }
    }
#pragma unroll
    for (int off = 1; off < 16; off <<= 1)
#pragma unroll
      for (int j = 0; j < 8; ++j)
        mloc[j] = fmaxf(mloc[j], __shfl_xor(mloc[j], off, 16));

    float alpha[8];
#pragma unroll
    for (int j = 0; j < 8; ++j) {
      const float mn = fmaxf(mrow[j], mloc[j]);
      alpha[j] = __expf(mrow[j] - mn);
      mrow[j] = mn;
    }

    float rs[8];
#pragma unroll
    for (int j = 0; j < 8; ++j) rs[j] = 0.0f;
#pragma unroll
    for (int nf = 0; nf < 4; ++nf) {
#pragma unroll
      for (int j = 0; j < 8; ++j) {
        const float p = __expf(s[nf][j] - mrow[j]);
        rs[j] += p;
        ldsP[wave][half * 8 + j][nf * 16 + r] = (__bf16)p;
      }
    }
#pragma unroll
    for (int off = 1; off < 16; off <<= 1)
#pragma unroll
      for (int j = 0; j < 8; ++j)
        rs[j] += __shfl_xor(rs[j], off, 16);
#pragma unroll
    for (int j = 0; j < 8; ++j) lrow[j] = lrow[j] * alpha[j] + rs[j];

#pragma unroll
    for (int df = 0; df < 4; ++df)
#pragma unroll
      for (int j = 0; j < 8; ++j) o[df][j] *= alpha[j];

#pragma unroll
    for (int ks = 0; ks < 2; ++ks) {
      const __bf16* pp = &ldsP[wave][r][ks * 32 + half * 16];
      v16bf pa = bf_cat(*(const v8bf*)pp, *(const v8bf*)(pp + 8));
#pragma unroll
      for (int df = 0; df < 4; ++df) {
        const int d = df * 16 + r;
        v16bf vf = *(const v16bf*)(vb + (size_t)d * SEQ + kt + ks * 32 + half * 16);
        o[df] = wmma_bf16(pa, vf, o[df]);
      }
    }
  }

#pragma unroll
  for (int df = 0; df < 4; ++df) {
    const int col = h * HDIM + df * 16 + r;
#pragma unroll
    for (int j = 0; j < 8; ++j) {
      const int t = q0 + half * 8 + j;
      y[((size_t)b * SEQ + t) * DMODEL + col] = (__bf16)(o[df][j] / lrow[j]);
    }
  }
}

// ---------------------------------------------------------------------------
extern "C" void kernel_launch(void* const* d_in, const int* in_sizes, int n_in,
                              void* d_out, int out_size, void* d_ws, size_t ws_size,
                              hipStream_t stream) {
  const float* x     = (const float*)d_in[0];
  const float* Wqkv  = (const float*)d_in[1];
  const float* bqkv  = (const float*)d_in[2];
  const float* Wproj = (const float*)d_in[3];
  const float* bproj = (const float*)d_in[4];
  float* out = (float*)d_out;

  char* w = (char*)d_ws;
  constexpr size_t NTOK      = (size_t)BATCH * SEQ;                      // 4096
  constexpr size_t QKV_BYTES = NTOK * 3 * DMODEL * sizeof(float);        // 50.3 MB
  constexpr size_t BHD_BYTES = NTOK * DMODEL * sizeof(__bf16);           // 8.4 MB

  float*  qkv    = (float*)w;                          w += QKV_BYTES;
  __bf16* qr     = (__bf16*)w;                         w += BHD_BYTES;
  __bf16* kr     = (__bf16*)w;                         w += BHD_BYTES;
  __bf16* vT     = (__bf16*)w;                         w += BHD_BYTES;
  __bf16* xb     = (__bf16*)w;                         w += BHD_BYTES;
  __bf16* yb     = (__bf16*)w;                         w += BHD_BYTES;
  __bf16* WqkvT  = (__bf16*)w;                         w += (size_t)3 * DMODEL * DMODEL * sizeof(__bf16);
  __bf16* WprojT = (__bf16*)w;

  const int M = (int)NTOK;  // 4096

  // 0) prep: bf16 conversions / weight transposes
  cvt_bf16<<<(M * DMODEL) / 256, 256, 0, stream>>>(x, xb, M * DMODEL);
  transpose_cvt<<<(DMODEL * 3 * DMODEL) / 256, 256, 0, stream>>>(Wqkv, WqkvT,
                                                                 DMODEL, 3 * DMODEL);
  transpose_cvt<<<(DMODEL * DMODEL) / 256, 256, 0, stream>>>(Wproj, WprojT,
                                                             DMODEL, DMODEL);

  // 1) qkv = x @ Wqkv + bqkv   [4096, 3072]
  gemm_bias_bf16<<<dim3((3 * DMODEL) / 128, M / 128), 256, 0, stream>>>(
      xb, WqkvT, bqkv, qkv, M, 3 * DMODEL, DMODEL);

  // 2) RoPE + pack to bf16
  rope_pack<<<(BATCH * SEQ * DMODEL) / 256, 256, 0, stream>>>(qkv, qr, kr, vT);

  // 3) flash attention -> yb [4096, 1024] bf16
  flash_attn<<<BATCH * NHEAD * (SEQ / 64), 128, 0, stream>>>(qr, kr, vT, yb);

  // 4) out = yb @ Wproj + bproj
  gemm_bias_bf16<<<dim3(DMODEL / 128, M / 128), 256, 0, stream>>>(
      yb, WprojT, bproj, out, M, DMODEL, DMODEL);
}
